// MoEDataRater_36558761624263
// MI455X (gfx1250) — compile-verified, matrix operand order checked by reading, not run on hardware
//
#include <hip/hip_runtime.h>
#include <hip/hip_bf16.h>

// Problem constants (from reference)
#define B_SZ   8192
#define S_SZ   64
#define H_SZ   320
#define E_SZ   4
#define K_SZ   2
#define NROUTE (B_SZ * K_SZ)            // 16384
#define CAPACITY 5120                   // ceil(1.25 * 8192 * 2 / 4)
#define EPSF   1e-9f
#define NKB    (H_SZ / 32)              // 10 K-steps of 32
#define NFT    8                        // 128 hidden units / 16 per wave

typedef __attribute__((ext_vector_type(16))) _Float16 v16h;
typedef __attribute__((ext_vector_type(8)))  float    v8f;

// ---------------------------------------------------------------------------
// Phase 1: masked mean-pool over S.  THE bandwidth phase: 671 MB streamed once
// at the 23.3 TB/s roofline (~29 us floor).  One block per batch row, 320
// threads (one per H column); each s-step reads one contiguous 1280B row.
// ---------------------------------------------------------------------------
__global__ __launch_bounds__(H_SZ)
void pool_kernel(const float* __restrict__ hidden,
                 const int*   __restrict__ mask,
                 float*       __restrict__ pooled) {
    const int b = blockIdx.x;
    const int h = threadIdx.x;
    const float* hp = hidden + (size_t)b * S_SZ * H_SZ + h;
    const int*   mp = mask + (size_t)b * S_SZ;
    float acc = 0.0f, ms = 0.0f;
    #pragma unroll 4
    for (int s = 0; s < S_SZ; ++s) {
        __builtin_prefetch(hp + (size_t)(s + 4) * H_SZ, 0, 0);  // global_prefetch_b8
        float m = (float)mp[s];
        acc += hp[(size_t)s * H_SZ] * m;
        ms  += m;
    }
    pooled[(size_t)b * H_SZ + h] = acc / fmaxf(ms, EPSF);
}

// ---------------------------------------------------------------------------
// Phase 2: router logits -> softmax -> top-2 (ties pick lower index, matching
// jax.lax.top_k), normalized top-k weights.  One thread per batch row.
// ---------------------------------------------------------------------------
__global__ __launch_bounds__(256)
void router_kernel(const float* __restrict__ pooled,
                   const float* __restrict__ rw,
                   const float* __restrict__ rb,
                   float* __restrict__ tvals,
                   int*   __restrict__ tidx,
                   float* __restrict__ tw) {
    const int b = blockIdx.x * blockDim.x + threadIdx.x;
    if (b >= B_SZ) return;
    const float* p = pooled + (size_t)b * H_SZ;
    float acc[E_SZ] = {0.f, 0.f, 0.f, 0.f};
    for (int h = 0; h < H_SZ; ++h) {
        float x = p[h];
        #pragma unroll
        for (int e = 0; e < E_SZ; ++e) acc[e] += x * rw[e * H_SZ + h];
    }
    float lg[E_SZ], pr[E_SZ];
    float mx = -INFINITY;
    #pragma unroll
    for (int e = 0; e < E_SZ; ++e) { lg[e] = acc[e] + rb[e]; mx = fmaxf(mx, lg[e]); }
    float sum = 0.f;
    #pragma unroll
    for (int e = 0; e < E_SZ; ++e) { pr[e] = __expf(lg[e] - mx); sum += pr[e]; }
    #pragma unroll
    for (int e = 0; e < E_SZ; ++e) pr[e] /= sum;

    int i0 = 0;
    #pragma unroll
    for (int e = 1; e < E_SZ; ++e) if (pr[e] > pr[i0]) i0 = e;   // strict > keeps lower idx
    int i1 = -1;
    #pragma unroll
    for (int e = 0; e < E_SZ; ++e) {
        if (e == i0) continue;
        if (i1 < 0 || pr[e] > pr[i1]) i1 = e;
    }
    const float v0 = pr[i0], v1 = pr[i1];
    const float den = fmaxf(v0 + v1, EPSF);
    tvals[b * 2 + 0] = v0;  tvals[b * 2 + 1] = v1;
    tidx [b * 2 + 0] = i0;  tidx [b * 2 + 1] = i1;
    tw   [b * 2 + 0] = v0 / den;  tw[b * 2 + 1] = v1 / den;
}

// ---------------------------------------------------------------------------
// Phase 3: capacity-limited greedy assignment without a sort.
// accepted_i = (#{j : e_j==e_i && (v_j>v_i || (v_j==v_i && j<i))} < CAPACITY)
// ---------------------------------------------------------------------------
#define ASSIGN_CHUNK 4096
__global__ __launch_bounds__(256)
void assign_kernel(const float* __restrict__ tvals,
                   const int*   __restrict__ tidx,
                   int*         __restrict__ accepted) {
    __shared__ float         sv[ASSIGN_CHUNK];
    __shared__ unsigned char se[ASSIGN_CHUNK];
    const int i  = blockIdx.x * blockDim.x + threadIdx.x;
    const float vi = tvals[i];
    const int   ei = tidx[i];
    int rank = 0;
    for (int cb = 0; cb < NROUTE; cb += ASSIGN_CHUNK) {
        __syncthreads();
        for (int t = threadIdx.x; t < ASSIGN_CHUNK; t += blockDim.x) {
            sv[t] = tvals[cb + t];
            se[t] = (unsigned char)tidx[cb + t];
        }
        __syncthreads();
        for (int t = 0; t < ASSIGN_CHUNK; ++t) {
            const int   j  = cb + t;
            const float vj = sv[t];
            const bool  win = (se[t] == (unsigned char)ei) &&
                              ((vj > vi) || ((vj == vi) && (j < i)));
            rank += win ? 1 : 0;
        }
    }
    accepted[i] = (rank < CAPACITY) ? 1 : 0;
}

// ---------------------------------------------------------------------------
// Phase 4a: pre-pack w1 (f32 [E,H,128]) into f16 B-fragments in the exact
// wave32 VGPR layout, so the GEMM loads each fragment as one 32B v16h.
// Storage index: (((e*NFT + nf)*NKB + kb)*32 + lane)*16 + i
// Element (lane,i) -> k = kb*32 + (lane>>4)*8 + ((i>>1)>=4?16:0)+((i>>1)&3)*2+(i&1)
//                     f = nf*16 + (lane&15)
// ---------------------------------------------------------------------------
__global__ __launch_bounds__(256)
void pack_b_kernel(const float* __restrict__ w1, _Float16* __restrict__ packedB) {
    const int idx = blockIdx.x * blockDim.x + threadIdx.x;   // 0 .. E*NFT*NKB*512-1
    const int i    = idx & 15;
    const int lane = (idx >> 4) & 31;
    int rest = idx >> 9;
    const int kb = rest % NKB;  rest /= NKB;
    const int nf = rest % NFT;
    const int e  = rest / NFT;
    const int k = kb * 32 + (lane >> 4) * 8 + (((i >> 1) >= 4) ? 16 : 0)
                + ((i >> 1) & 3) * 2 + (i & 1);
    const int f = nf * 16 + (lane & 15);
    packedB[idx] = (_Float16)w1[((size_t)e * H_SZ + k) * 128 + f];
}

// ---------------------------------------------------------------------------
// Phase 4b: dense expert MLPs via v_wmma_f32_16x16x32_f16, K fully unrolled
// (10 back-to-back WMMAs to cover the 5-slot RAW shadow).
// A tile (16x320 f32, one contiguous 20KB run of `pooled`) is pulled into LDS
// with global_load_async_to_lds_b128 (ASYNCcnt path), then converted to f16.
// grid = (B/16, E); 256 threads = 8 waves, each wave owns 16 of 128 f-columns.
// Epilogue: relu(c + b1) * w2 reduced over f via ds_add_f32.
// ---------------------------------------------------------------------------
__global__ __launch_bounds__(256)
void expert_gemm_kernel(const float* __restrict__ pooled,
                        const _Float16* __restrict__ packedB,
                        const float* __restrict__ b1,
                        const float* __restrict__ w2,
                        const float* __restrict__ b2,
                        float* __restrict__ expert_out) {
    __shared__ float    ldsAf32[16 * H_SZ];   // 20 KB, async-load target
    __shared__ _Float16 ldsA[16 * H_SZ];      // 10 KB, f16 A tile
    __shared__ float    ldsAcc[16];

    const int tileM = blockIdx.x;   // 0..511
    const int e     = blockIdx.y;   // 0..3
    const int tid   = threadIdx.x;

    // --- async copy: 20 KB contiguous (16 rows x 320 cols f32) -> LDS ------
    {
        const unsigned lbase = (unsigned)(size_t)(void*)ldsAf32;  // LDS offset in low 32b
        const float* gbase = pooled + (size_t)tileM * 16 * H_SZ;
        #pragma unroll
        for (int it = 0; it < 5; ++it) {                 // 1280 x b128 / 256 thr
            const int q = tid + it * 256;                // float4 index
            const float* g = gbase + q * 4;
            const unsigned laddr = lbase + q * 16;
            asm volatile("global_load_async_to_lds_b128 %0, %1, off"
                         :: "v"(laddr), "v"(g) : "memory");
        }
        asm volatile("s_wait_asynccnt 0x0" ::: "memory");
    }
    if (tid < 16) ldsAcc[tid] = 0.0f;
    __syncthreads();

    // --- convert A tile f32 -> f16 in LDS ----------------------------------
    for (int idx = tid; idx < 16 * H_SZ; idx += 256)
        ldsA[idx] = (_Float16)ldsAf32[idx];
    __syncthreads();

    const int wave  = tid >> 5;       // 0..7
    const int lane  = tid & 31;
    const int half  = lane >> 4;
    const int mn    = lane & 15;      // A: row M ; B/C: col N
    const int fcol  = wave * 16 + mn; // hidden-unit column this lane produces

    const v16h* bp = (const v16h*)packedB + ((size_t)(e * NFT + wave) * NKB) * 32 + lane;

    v8f c = {};
    #pragma unroll
    for (int kbi = 0; kbi < NKB; ++kbi) {
        const int koff = kbi * 32 + half * 8;
        v16h a;
        #pragma unroll
        for (int r = 0; r < 8; ++r) {       // two contiguous 16B runs -> 2x ds_load_b128
            const int k0 = koff + ((r >= 4) ? 16 : 0) + (r & 3) * 2;
            a[2 * r]     = ldsA[mn * H_SZ + k0];
            a[2 * r + 1] = ldsA[mn * H_SZ + k0 + 1];
        }
        const v16h bf = bp[kbi * 32];       // one 32B fragment load
        c = __builtin_amdgcn_wmma_f32_16x16x32_f16(
                false, a, false, bf, (short)0, c, false, false);
    }

    // Epilogue: C layout -> (m = half*8 + r, n = mn); f index == n.
    const float b1v = b1[e * 128 + fcol];
    const float w2v = w2[e * 128 + fcol];
    #pragma unroll
    for (int r = 0; r < 8; ++r) {
        const int   m = half * 8 + r;
        const float v = fmaxf(c[r] + b1v, 0.0f) * w2v;
        atomicAdd(&ldsAcc[m], v);                    // ds_add_f32
    }
    __syncthreads();
    if (tid < 16) {
        expert_out[(size_t)(tileM * 16 + tid) * E_SZ + e] = ldsAcc[tid] + b2[e];
    }
}

// ---------------------------------------------------------------------------
// Phase 5: weighted merge of accepted routes.
// ---------------------------------------------------------------------------
__global__ __launch_bounds__(256)
void merge_kernel(const float* __restrict__ tw,
                  const int*   __restrict__ tidx,
                  const int*   __restrict__ accepted,
                  const float* __restrict__ expert_out,
                  float*       __restrict__ scores) {
    const int b = blockIdx.x * blockDim.x + threadIdx.x;
    if (b >= B_SZ) return;
    const float a0 = tw[b * 2 + 0] * (float)accepted[b * 2 + 0];
    const float a1 = tw[b * 2 + 1] * (float)accepted[b * 2 + 1];
    const float den = fmaxf(a0 + a1, EPSF);
    const int e0 = tidx[b * 2 + 0];
    const int e1 = tidx[b * 2 + 1];
    scores[b] = (a0 * expert_out[(size_t)b * E_SZ + e0] +
                 a1 * expert_out[(size_t)b * E_SZ + e1]) / den;
}

// ---------------------------------------------------------------------------
extern "C" void kernel_launch(void* const* d_in, const int* in_sizes, int n_in,
                              void* d_out, int out_size, void* d_ws, size_t ws_size,
                              hipStream_t stream) {
    const float* hidden = (const float*)d_in[0];
    const int*   mask   = (const int*)  d_in[1];
    const float* rw     = (const float*)d_in[2];
    const float* rb     = (const float*)d_in[3];
    const float* w1     = (const float*)d_in[4];
    const float* b1     = (const float*)d_in[5];
    const float* w2     = (const float*)d_in[6];
    const float* b2     = (const float*)d_in[7];
    float* scores = (float*)d_out;

    char* ws = (char*)d_ws;
    float*    pooled = (float*)ws;     ws += (size_t)B_SZ * H_SZ * sizeof(float);
    _Float16* packB  = (_Float16*)ws;  ws += (size_t)E_SZ * NFT * NKB * 512 * sizeof(_Float16);
    float*    tvals  = (float*)ws;     ws += (size_t)NROUTE * sizeof(float);
    int*      tidx   = (int*)  ws;     ws += (size_t)NROUTE * sizeof(int);
    float*    tw     = (float*)ws;     ws += (size_t)NROUTE * sizeof(float);
    int*      acc    = (int*)  ws;     ws += (size_t)NROUTE * sizeof(int);
    float*    eout   = (float*)ws;     ws += (size_t)B_SZ * E_SZ * sizeof(float);

    pool_kernel<<<B_SZ, H_SZ, 0, stream>>>(hidden, mask, pooled);
    pack_b_kernel<<<(E_SZ * NFT * NKB * 512) / 256, 256, 0, stream>>>(w1, packB);
    router_kernel<<<B_SZ / 256, 256, 0, stream>>>(pooled, rw, rb, tvals, tidx, tw);
    assign_kernel<<<NROUTE / 256, 256, 0, stream>>>(tvals, tidx, acc);
    dim3 g(B_SZ / 16, E_SZ);
    expert_gemm_kernel<<<g, 256, 0, stream>>>(pooled, packB, b1, w2, b2, eout);
    merge_kernel<<<B_SZ / 256, 256, 0, stream>>>(tw, tidx, acc, eout, scores);
}
